// MoE_5308579577948
// MI455X (gfx1250) — compile-verified, matrix operand order checked by reading, not run on hardware
//
#include <hip/hip_runtime.h>
#include <hip/hip_bf16.h>

// ---------------- problem constants ----------------
constexpr int L   = 256;   // sequence length
constexpr int Bsz = 128;   // batch
constexpr int E   = 512;   // embed dim
constexpr int NE  = 8;     // experts
constexpr int H   = 8;     // heads
constexpr int S   = 256;   // key length (== L)
constexpr int JOBS = Bsz * 2;          // (batch, slot) pairs, top-2 routing
constexpr size_t OUT_ELEMS = (size_t)L * Bsz * E;       // combined_out
constexpr size_t W_ELEMS   = (size_t)Bsz * L * S;       // combined_w

typedef __attribute__((ext_vector_type(16))) _Float16 v16h;
typedef __attribute__((ext_vector_type(8)))  float    v8f;

union FragH { v16h v; _Float16 h[16]; };
union FragF { v8f  v; float     f[8];  };

#define DEV __device__ __forceinline__

// 16-bit A-matrix 16x32 per-lane K map (ISA 7.12.2); identical map reused for B
// staged as [n][k] tiles.
DEV int kmap(int i, int hv) {
  int p = i >> 1, bit = i & 1;
  int base = (p < 4) ? (p << 1) : (16 + ((p - 4) << 1));
  return base + bit + (hv << 3);
}

DEV v16h load_frag(const _Float16* rowbase, int hv) {
  FragH a;
#pragma unroll
  for (int i = 0; i < 16; ++i) a.h[i] = rowbase[kmap(i, hv)];
  return a.v;
}

DEV v8f wmma_f16(v16h a, v16h b, v8f c) {
  return __builtin_amdgcn_wmma_f32_16x16x32_f16(false, a, false, b,
                                                (short)0, c, false, false);
}

// ---- CDNA5 async global->LDS copy (ASYNCcnt-tracked, 16B per lane) ----------
DEV void async_ld_b128(unsigned lds_off, const _Float16* gaddr) {
  asm volatile("global_load_async_to_lds_b128 %0, %1, off"
               :: "v"(lds_off), "v"(gaddr)
               : "memory");
}
DEV void wait_async0() {
  asm volatile("s_wait_asynccnt 0x0" ::: "memory");
}
DEV unsigned lds_off32(const void* p) {        // low 32 bits = LDS byte offset
  return (unsigned)(unsigned long long)p;
}

// ---------------- deterministic hash Gaussian ----------------
DEV unsigned hashu(unsigned x) {
  x ^= x >> 16; x *= 0x7feb352du;
  x ^= x >> 15; x *= 0x846ca68bu;
  x ^= x >> 16; return x;
}
DEV float rng_normal(unsigned i) {
  unsigned a = hashu(i * 2654435761u + 0x9e3779b9u);
  unsigned c = hashu(i * 2246822519u + 0x85ebca6bu);
  float u1 = ((float)a + 0.5f) * (1.0f / 4294967296.0f);
  float u2 = ((float)c + 0.5f) * (1.0f / 4294967296.0f);
  return sqrtf(-2.0f * logf(u1)) * cosf(6.28318530718f * u2);
}

// ---------------- gating ----------------
__global__ __launch_bounds__(256) void k_qsum(const float* __restrict__ q,
                                              float* __restrict__ qsum) {
  int idx = blockIdx.x * 256 + threadIdx.x;          // idx = b*E + e, < Bsz*E
  float s = 0.f;
  for (int l = 0; l < L; ++l) s += q[(size_t)l * Bsz * E + idx];
  qsum[idx] = s;
}

__global__ __launch_bounds__(256) void k_gate(const float* __restrict__ qsum,
                                              const float* __restrict__ w_gate,
                                              const float* __restrict__ w_noise,
                                              float* __restrict__ gates,
                                              float* __restrict__ probb,
                                              int* __restrict__ tidx) {
  __shared__ float cl[NE], rw[NE];
  int b = blockIdx.x;
  int t = threadIdx.x, f = t >> 5, lane = t & 31;
  float sc = 0.f, sn = 0.f;
  for (int k = lane; k < E; k += 32) {
    float qv = qsum[b * E + k];
    sc += qv * w_gate[k * NE + f];
    sn += qv * w_noise[k * NE + f];
  }
#pragma unroll
  for (int off = 16; off; off >>= 1) {
    sc += __shfl_down(sc, off);
    sn += __shfl_down(sn, off);
  }
  if (lane == 0) { cl[f] = sc; rw[f] = sn; }
  __syncthreads();
  if (t == 0) {
    float clean[NE], sd[NE], noisy[NE];
    for (int e = 0; e < NE; ++e) {
      clean[e] = cl[e];
      float r = rw[e];
      float sp = (r > 20.f) ? r : log1pf(expf(r));   // softplus
      sd[e] = sp + 0.01f;
      noisy[e] = clean[e] + rng_normal((unsigned)(b * NE + e)) * sd[e];
    }
    // top-3 of 8
    int i0 = 0, i1 = 0, i2 = 0;
    float v0 = -1e30f, v1 = -1e30f, v2 = -1e30f;
    for (int e = 0; e < NE; ++e) {
      float x = noisy[e];
      if (x > v0)      { v2 = v1; i2 = i1; v1 = v0; i1 = i0; v0 = x; i0 = e; }
      else if (x > v1) { v2 = v1; i2 = i1; v1 = x;  i1 = e; }
      else if (x > v2) { v2 = x;  i2 = e; }
    }
    (void)i2;
    float e1 = expf(v1 - v0);
    float g0 = 1.0f / (1.0f + e1);
    float g1 = e1 / (1.0f + e1);
    for (int e = 0; e < NE; ++e) gates[b * NE + e] = 0.f;
    gates[b * NE + i0] = g0;
    gates[b * NE + i1] = g1;
    tidx[2 * b]     = i0;
    tidx[2 * b + 1] = i1;
    float thr_in = v2, thr_out = v1;
    for (int e = 0; e < NE; ++e) {
      float thr = (noisy[e] > thr_in) ? thr_in : thr_out;
      float z = (clean[e] - thr) / sd[e];
      probb[b * NE + e] = 0.5f * erfcf(-z * 0.70710678118f);  // normal CDF
    }
  }
}

__global__ void k_loss(const float* __restrict__ gates,
                       const float* __restrict__ probb,
                       float* __restrict__ loss_out) {
  __shared__ float imp[NE], ld[NE];
  int t = threadIdx.x;
  if (t < NE) {
    float si = 0.f, sl = 0.f;
    for (int b = 0; b < Bsz; ++b) {
      si += gates[b * NE + t];
      sl += probb[b * NE + t];
    }
    imp[t] = si; ld[t] = sl;
  }
  __syncthreads();
  if (t == 0) {
    auto cv2 = [](const float* x) {
      float m = 0.f;
      for (int i = 0; i < NE; ++i) m += x[i];
      m *= (1.0f / NE);
      float v = 0.f;
      for (int i = 0; i < NE; ++i) v += (x[i] - m) * (x[i] - m);
      v *= (1.0f / (NE - 1));                       // ddof=1
      return v / (m * m + 1e-10f);
    };
    loss_out[0] = (cv2(imp) + cv2(ld)) * 0.01f;
  }
}

__global__ __launch_bounds__(256) void k_zero(float* __restrict__ p, int n) {
  int i = blockIdx.x * 256 + threadIdx.x;
  if (i < n) p[i] = 0.f;
}

// ---------------- indexed QKV projection: Y = (X @ W^T + b) [*scale] ----------
__global__ __launch_bounds__(256) void k_proj(const float* __restrict__ query,
                                              const float* __restrict__ key_t,
                                              const float* __restrict__ value,
                                              const float* __restrict__ W_in,
                                              const float* __restrict__ b_in,
                                              const int* __restrict__ tidx,
                                              _Float16* __restrict__ qh,
                                              _Float16* __restrict__ kh,
                                              _Float16* __restrict__ vh) {
  __shared__ _Float16 As[64 * 34];
  __shared__ _Float16 Bs[64 * 34];
  int bid = blockIdx.x;
  int nt = bid & 7, mt = (bid >> 3) & 3, rest = bid >> 5;
  int which = rest % 3, j = rest / 3;
  int b = j >> 1, slot = j & 1;
  int e = tidx[2 * b + slot];
  const float* X = (which == 0) ? query : (which == 1) ? key_t : value;
  _Float16* Y = (which == 0) ? qh : (which == 1) ? kh : vh;
  const float* W = W_in + ((size_t)e * 3 * E + (size_t)which * E) * E;
  const float* bias = b_in + (size_t)e * 3 * E + which * E;
  int m0 = mt * 64, n0 = nt * 64;
  int t = threadIdx.x, lane = t & 31, wv = t >> 5;
  int hv = lane >> 4, l16 = lane & 15;
  int mi = wv >> 1, nj0 = (wv & 1) * 2, nj1 = nj0 + 1;
  FragF c0, c1;
#pragma unroll
  for (int i = 0; i < 8; ++i) { c0.f[i] = 0.f; c1.f[i] = 0.f; }
  for (int k0 = 0; k0 < E; k0 += 32) {
    // prefetch next K tile one step ahead (global_prefetch_b8)
    if (k0 + 32 < E) {
      int m = t >> 2;                                    // 0..63
      __builtin_prefetch(&X[((size_t)(m0 + m) * Bsz + b) * E + k0 + 32], 0, 1);
      __builtin_prefetch(&W[(size_t)(n0 + m) * E + k0 + 32], 0, 1);
    }
#pragma unroll
    for (int r = 0; r < 8; ++r) {
      int u = t + r * 256, m = u >> 5, k = u & 31;
      As[m * 34 + k] = (_Float16)X[((size_t)(m0 + m) * Bsz + b) * E + k0 + k];
      Bs[m * 34 + k] = (_Float16)W[(size_t)(n0 + m) * E + k0 + k];
    }
    __syncthreads();
    v16h a  = load_frag(&As[(mi * 16 + l16) * 34], hv);
    v16h b0 = load_frag(&Bs[(nj0 * 16 + l16) * 34], hv);
    v16h b1 = load_frag(&Bs[(nj1 * 16 + l16) * 34], hv);
    c0.v = wmma_f16(a, b0, c0.v);
    c1.v = wmma_f16(a, b1, c1.v);
    __syncthreads();
  }
  float scale = (which == 0) ? 0.125f : 1.0f;   // hd^-0.5 fused into qh
  int nA = n0 + nj0 * 16 + l16, nB = n0 + nj1 * 16 + l16;
  float bA = bias[nA], bB = bias[nB];
#pragma unroll
  for (int v = 0; v < 8; ++v) {
    int m = m0 + mi * 16 + v + 8 * hv;
    Y[((size_t)j * L + m) * E + nA] = (_Float16)((c0.f[v] + bA) * scale);
    Y[((size_t)j * L + m) * E + nB] = (_Float16)((c1.f[v] + bB) * scale);
  }
}

// ---------------- attention per (job, head, 32-row tile) ----------------------
// Q/K tiles are already f16 in workspace -> staged with
// GLOBAL_LOAD_ASYNC_TO_LDS_B128 (ASYNCcnt), 16B-aligned LDS rows (stride 72).
__global__ __launch_bounds__(256) void k_attn(const _Float16* __restrict__ qh,
                                              const _Float16* __restrict__ kh,
                                              const _Float16* __restrict__ vh,
                                              _Float16* __restrict__ ao,
                                              float* __restrict__ wsum) {
  __shared__ _Float16 Qs[32 * 72];      // 72 halfs = 144B rows (16B aligned)
  __shared__ float    Sc[32 * 257];
  __shared__ _Float16 Ks[64 * 72];
  __shared__ _Float16 Vt[64 * 34];
  int bid = blockIdx.x;
  int mt = bid & 7, h = (bid >> 3) & 7, j = bid >> 6;
  int m0 = mt * 32, ho = h * 64;
  int t = threadIdx.x, lane = t & 31, wv = t >> 5;
  int hv = lane >> 4, l16 = lane & 15;
  // stage Q: 32 rows x 64 halfs = 256 x 16B chunks, one async per thread
  {
    int m = t >> 3, c = t & 7;
    async_ld_b128(lds_off32(&Qs[m * 72 + c * 8]),
                  &qh[((size_t)j * L + m0 + m) * E + ho + c * 8]);
  }
  // scores = qh @ kh^T   (K = hd = 64)
  {
    int mi = wv & 1, sj = wv >> 1;
    for (int s0 = 0; s0 < S; s0 += 64) {
      __syncthreads();                      // readers of previous Ks done
#pragma unroll
      for (int r = 0; r < 2; ++r) {         // 64 rows x 64 halfs = 512 chunks
        int u = t + r * 256, n = u >> 3, c = u & 7;
        async_ld_b128(lds_off32(&Ks[n * 72 + c * 8]),
                      &kh[((size_t)j * L + s0 + n) * E + ho + c * 8]);
      }
      wait_async0();
      __syncthreads();
      FragF c;
#pragma unroll
      for (int i = 0; i < 8; ++i) c.f[i] = 0.f;
#pragma unroll
      for (int kk = 0; kk < 64; kk += 32) {
        v16h a  = load_frag(&Qs[(mi * 16 + l16) * 72 + kk], hv);
        v16h bf = load_frag(&Ks[(sj * 16 + l16) * 72 + kk], hv);
        c.v = wmma_f16(a, bf, c.v);
      }
#pragma unroll
      for (int v = 0; v < 8; ++v) {
        int m = mi * 16 + v + 8 * hv;
        Sc[m * 257 + s0 + sj * 16 + l16] = c.f[v];
      }
    }
  }
  __syncthreads();
  // row softmax; overwrite row in place with f16 probs (safe: halves trail f32)
  if (t < 32) {
    float* row = &Sc[t * 257];
    float mx = -1e30f;
    for (int s = 0; s < S; ++s) mx = fmaxf(mx, row[s]);
    float sum = 0.f;
    for (int s = 0; s < S; ++s) sum += expf(row[s] - mx);
    float inv = 1.0f / sum;
    _Float16* pr = (_Float16*)row;
    for (int s = 0; s < S; ++s) pr[s] = (_Float16)(expf(row[s] - mx) * inv);
  }
  __syncthreads();
  // accumulate head-mean attention weights
  for (int u = t; u < 32 * S; u += 256) {
    int m = u >> 8, s = u & 255;
    float p = (float)((const _Float16*)&Sc[m * 257])[s];
    atomicAdd(&wsum[((size_t)j * L + m0 + m) * S + s], p * (1.0f / H));
  }
  // out = probs @ vh   (K = S = 256); V staged transposed (scalar: layout swap)
  {
    int mi = wv >> 2, ni = wv & 3;
    FragF c;
#pragma unroll
    for (int i = 0; i < 8; ++i) c.f[i] = 0.f;
    for (int k0 = 0; k0 < S; k0 += 32) {
      __syncthreads();
#pragma unroll
      for (int r = 0; r < 8; ++r) {
        int u = t + r * 256, n = u >> 5, k = u & 31;
        Vt[n * 34 + k] = vh[((size_t)j * L + k0 + k) * E + ho + n];
      }
      __syncthreads();
      const _Float16* prow = (const _Float16*)&Sc[(mi * 16 + l16) * 257];
      v16h a  = load_frag(prow + k0, hv);
      v16h bf = load_frag(&Vt[(ni * 16 + l16) * 34], hv);
      c.v = wmma_f16(a, bf, c.v);
    }
#pragma unroll
    for (int v = 0; v < 8; ++v) {
      int m = m0 + mi * 16 + v + 8 * hv;
      int n = ho + ni * 16 + l16;
      ao[((size_t)j * L + m) * E + n] = (_Float16)c.f[v];
    }
  }
}

// ---------------- out-projection + top-2 exp/log combine ---------------------
__global__ __launch_bounds__(256) void k_outproj(const _Float16* __restrict__ ao,
                                                 const float* __restrict__ W_out,
                                                 const float* __restrict__ b_out,
                                                 const int* __restrict__ tidx,
                                                 float* __restrict__ out) {
  __shared__ _Float16 As[64 * 40];      // 40 halfs = 80B rows (16B aligned)
  __shared__ _Float16 Bs[64 * 34];
  int bid = blockIdx.x;
  int nt = bid & 7, mt = (bid >> 3) & 3, b = bid >> 5;
  int m0 = mt * 64, n0 = nt * 64;
  int t = threadIdx.x, lane = t & 31, wv = t >> 5;
  int hv = lane >> 4, l16 = lane & 15;
  int mi = wv >> 1, nj0 = (wv & 1) * 2, nj1 = nj0 + 1;
  int nA = n0 + nj0 * 16 + l16, nB = n0 + nj1 * 16 + l16;
  FragF acc[2][2];
  for (int slot = 0; slot < 2; ++slot) {
    int j = 2 * b + slot;
    int e = tidx[j];
    const float* Wo = W_out + (size_t)e * E * E;
    const float* bo = b_out + (size_t)e * E;
    FragF c0, c1;
#pragma unroll
    for (int i = 0; i < 8; ++i) { c0.f[i] = 0.f; c1.f[i] = 0.f; }
    for (int k0 = 0; k0 < E; k0 += 32) {
      __syncthreads();
      {  // A tile: 64 rows x 32 halfs = 256 x 16B chunks, async to LDS
        int m = t >> 2, c = t & 3;
        async_ld_b128(lds_off32(&As[m * 40 + c * 8]),
                      &ao[((size_t)j * L + m0 + m) * E + k0 + c * 8]);
      }
#pragma unroll
      for (int r = 0; r < 8; ++r) {   // W tile: f32 -> f16 convert (VALU path)
        int u = t + r * 256, m = u >> 5, k = u & 31;
        Bs[m * 34 + k] = (_Float16)Wo[(size_t)(n0 + m) * E + k0 + k];
      }
      wait_async0();
      __syncthreads();
      v16h a   = load_frag(&As[(mi * 16 + l16) * 40], hv);
      v16h bb0 = load_frag(&Bs[(nj0 * 16 + l16) * 34], hv);
      v16h bb1 = load_frag(&Bs[(nj1 * 16 + l16) * 34], hv);
      c0.v = wmma_f16(a, bb0, c0.v);
      c1.v = wmma_f16(a, bb1, c1.v);
    }
    float bA = bo[nA], bB = bo[nB];
#pragma unroll
    for (int v = 0; v < 8; ++v) { c0.f[v] += bA; c1.f[v] += bB; }
    acc[slot][0] = c0;
    acc[slot][1] = c1;
  }
#pragma unroll
  for (int v = 0; v < 8; ++v) {
    int m = m0 + mi * 16 + v + 8 * hv;
    out[((size_t)m * Bsz + b) * E + nA] =
        logf(expf(acc[0][0].f[v]) + expf(acc[1][0].f[v]));
    out[((size_t)m * Bsz + b) * E + nB] =
        logf(expf(acc[0][1].f[v]) + expf(acc[1][1].f[v]));
  }
}

__global__ __launch_bounds__(256) void k_wcomb(const float* __restrict__ wsum,
                                               float* __restrict__ outw) {
  size_t g = (size_t)blockIdx.x * 256 + threadIdx.x;   // < Bsz*L*S
  size_t ls = (size_t)L * S;
  size_t b = g / ls, r = g % ls;
  float a = wsum[(2 * b) * ls + r];
  float c = wsum[(2 * b + 1) * ls + r];
  outw[g] = logf(expf(a) + expf(c));
}

// ---------------- host side ----------------
extern "C" void kernel_launch(void* const* d_in, const int* in_sizes, int n_in,
                              void* d_out, int out_size, void* d_ws, size_t ws_size,
                              hipStream_t stream) {
  (void)in_sizes; (void)n_in; (void)out_size; (void)ws_size;
  const float* query   = (const float*)d_in[0];
  const float* key_t   = (const float*)d_in[1];
  const float* value   = (const float*)d_in[2];
  const float* w_gate  = (const float*)d_in[3];
  const float* w_noise = (const float*)d_in[4];
  const float* W_in    = (const float*)d_in[5];
  const float* b_in    = (const float*)d_in[6];
  const float* W_out   = (const float*)d_in[7];
  const float* b_out   = (const float*)d_in[8];
  float* out = (float*)d_out;

  char* ws = (char*)d_ws;
  size_t off = 0;
  auto alloc = [&](size_t bytes) -> void* {
    void* p = ws + off;
    off = (off + bytes + 255) & ~(size_t)255;
    return p;
  };
  float*    qsum  = (float*)alloc((size_t)Bsz * E * 4);
  float*    gates = (float*)alloc((size_t)Bsz * NE * 4);
  float*    probb = (float*)alloc((size_t)Bsz * NE * 4);
  int*      tidx  = (int*)  alloc((size_t)Bsz * 2 * 4);
  _Float16* qh = (_Float16*)alloc((size_t)JOBS * L * E * 2);
  _Float16* kh = (_Float16*)alloc((size_t)JOBS * L * E * 2);
  _Float16* vh = (_Float16*)alloc((size_t)JOBS * L * E * 2);
  _Float16* ao = (_Float16*)alloc((size_t)JOBS * L * E * 2);
  float*  wsum = (float*)   alloc((size_t)JOBS * L * S * 4);

  // 1) gating
  k_qsum<<<(Bsz * E) / 256, 256, 0, stream>>>(query, qsum);
  k_gate<<<Bsz, 256, 0, stream>>>(qsum, w_gate, w_noise, gates, probb, tidx);
  k_loss<<<1, 64, 0, stream>>>(gates, probb, out + OUT_ELEMS);
  // 2) zero weight accumulator (atomics target; must re-zero every call)
  k_zero<<<(JOBS * L * S) / 256, 256, 0, stream>>>(wsum, JOBS * L * S);
  // 3) indexed QKV projections: jobs * {q,k,v} * (L/64) * (E/64) blocks
  k_proj<<<JOBS * 3 * (L / 64) * (E / 64), 256, 0, stream>>>(
      query, key_t, value, W_in, b_in, tidx, qh, kh, vh);
  // 4) attention: jobs * H * (L/32) blocks
  k_attn<<<JOBS * H * (L / 32), 256, 0, stream>>>(qh, kh, vh, ao, wsum);
  // 5) out-projection + top-2 combine straight into d_out
  k_outproj<<<Bsz * (L / 64) * (E / 64), 256, 0, stream>>>(
      ao, W_out, b_out, tidx, out);
  // 6) combine attention weights
  k_wcomb<<<(int)(W_ELEMS / 256), 256, 0, stream>>>(wsum, out + OUT_ELEMS + 1);
}
